// LSTM_39934605918623
// MI455X (gfx1250) — matrix operand order repaired and checked by reading, not compiled, not run on hardware
//
#include <hip/hip_runtime.h>
#include <hip/hip_bf16.h>

// ---------------------------------------------------------------------------
// 2-layer LSTM for MI455X (gfx1250), wave32 + WMMA bf16 split-precision GEMMs.
//   gates = [x | h] * [W | R]^T  (K-concat), f32 accumulate via
//   v_wmma_f32_16x16x32_bf16 with hi/lo bf16 split (3 WMMAs per K=32 chunk).
// ---------------------------------------------------------------------------

typedef __attribute__((ext_vector_type(16))) __bf16 v16bf;
typedef __attribute__((ext_vector_type(8)))  __bf16 v8bf;
typedef __attribute__((ext_vector_type(8)))  float  v8f;

#define SEQL  128
#define DIN   768
#define HID   512
#define NCLS  9
#define BATCH 256
#define KC    128     // K-chunk staged in LDS per iteration
#define AST   136     // padded LDS row stride in bf16 elems (272B -> no bank conflicts)

__device__ __forceinline__ float sigmoidf_(float x) { return 1.0f / (1.0f + __expf(-x)); }

__device__ __forceinline__ v16bf cat8(v8bf a, v8bf b) {
  return __builtin_shufflevector(a, b, 0,1,2,3,4,5,6,7,8,9,10,11,12,13,14,15);
}

// ---------------------------------------------------------------------------
// Fused LSTM cell: GEMM (all 4 gates) + bias + nonlinearity + state update.
// mode 0: A = [embed(x_t) | h1_prev], K=1280   (uses ids/tok/pos + a2)
// mode 1: A = [h1_new     | h2_prev], K=1024   (uses a1 + a2)
// Block = 128 thr (4 waves). Wave w owns batch-tile b0..b0+15 x hidden cols
// h0..h0+15 (h0 = blockIdx.y*64 + w*16) for ALL 4 gates -> in-wave combine.
// ---------------------------------------------------------------------------
__global__ __launch_bounds__(128) void lstm_cell(
    int mode, int t,
    const int*   __restrict__ ids, const float* __restrict__ tok,
    const float* __restrict__ pos,
    const float* __restrict__ a1,      // mode1: first 512 K-cols source
    const float* __restrict__ a2,      // last 512 K-cols source (h_prev)
    const __bf16* __restrict__ Wh, const __bf16* __restrict__ Wl,  // [2048][Ktot] hi/lo
    const float* __restrict__ bias,    // [4][512]
    float* __restrict__ Cst,           // in-place cell state [256][512]
    float* __restrict__ Hnew,          // out hidden state   [256][512]
    int Ktot)
{
  __shared__ __align__(16) __bf16 Ah[2][16 * AST];
  __shared__ __align__(16) __bf16 Al[2][16 * AST];

  const int tid  = threadIdx.x;
  const int lane = tid & 31;
  const int wv   = tid >> 5;
  const int b0   = blockIdx.x * 16;
  const int h0   = blockIdx.y * 64 + wv * 16;
  const int K1   = Ktot - HID;
  const int NKC  = Ktot / KC;

  const v8f vzero = {0.f,0.f,0.f,0.f,0.f,0.f,0.f,0.f};
  v8f acc[4];
  #pragma unroll
  for (int g = 0; g < 4; ++g) acc[g] = vzero;

  // ISA fragment addressing (wave32):
  //  A (16x32 bf16): lanes 0-15 row M=lane, K = base 0..7 & 16..23; lanes 16-31 base 8.
  //  B (32x16 bf16): lanes 0-15 col N=lane, K = 0..7 & 16..23; lanes 16-31 base 8.
  const int mloc = lane & 15;
  const int akb  = (lane < 16) ? 0 : 8;
  const int bkb  = (lane < 16) ? 0 : 8;
  const int ncol = h0 + mloc;

  for (int kc = 0; kc < NKC; ++kc) {
    const int buf = kc & 1;
    // --- stage 16 x KC fp32 -> bf16 hi/lo split into LDS (coalesced) ---
    for (int e = tid; e < 16 * KC; e += 128) {
      const int m  = e >> 7;
      const int k  = e & (KC - 1);
      const int kg = kc * KC + k;
      const int b  = b0 + m;
      float v;
      if (kg < K1) {
        if (mode == 0) {
          const int tk = ids[(size_t)b * SEQL + t];
          v = tok[(size_t)tk * DIN + kg] + pos[(size_t)t * DIN + kg];
        } else {
          v = a1[(size_t)b * HID + kg];
        }
      } else {
        v = a2[(size_t)b * HID + (kg - K1)];
      }
      const __bf16 hi = (__bf16)v;
      Ah[buf][m * AST + k] = hi;
      Al[buf][m * AST + k] = (__bf16)(v - (float)hi);
    }
    __syncthreads();   // one barrier/iter is safe with 2-deep LDS buffering

    // --- 4 K-sub-chunks of 32, 4 gates, 3-term split WMMA each ---
    #pragma unroll
    for (int s = 0; s < 4; ++s) {
      const __bf16* ap  = &Ah[buf][mloc * AST + s * 32 + akb];
      const __bf16* alp = &Al[buf][mloc * AST + s * 32 + akb];
      const v16bf afh = cat8(*(const v8bf*)ap,  *(const v8bf*)(ap  + 16));
      const v16bf afl = cat8(*(const v8bf*)alp, *(const v8bf*)(alp + 16));
      #pragma unroll
      for (int g = 0; g < 4; ++g) {
        const size_t nrow = (size_t)(g * HID + ncol);
        const __bf16* qh = Wh + nrow * (size_t)Ktot + kc * KC + s * 32 + bkb;
        const __bf16* ql = Wl + nrow * (size_t)Ktot + kc * KC + s * 32 + bkb;
        if (s == 0 && kc + 1 < NKC) __builtin_prefetch(qh + KC, 0, 1);  // next chunk (L2-resident)
        const v16bf bfh = cat8(*(const v8bf*)qh, *(const v8bf*)(qh + 16));
        const v16bf bfl = cat8(*(const v8bf*)ql, *(const v8bf*)(ql + 16));
        acc[g] = __builtin_amdgcn_wmma_f32_16x16x32_bf16(false, afh, false, bfh, (short)0, acc[g], false, false);
        acc[g] = __builtin_amdgcn_wmma_f32_16x16x32_bf16(false, afl, false, bfh, (short)0, acc[g], false, false);
        acc[g] = __builtin_amdgcn_wmma_f32_16x16x32_bf16(false, afh, false, bfl, (short)0, acc[g], false, false);
      }
    }
  }

  // --- in-wave gate combine + state update ---
  // C/D layout: vgpr r -> row M = r (lanes 0-15) or 8+r (lanes 16-31), col N = lane%16.
  const float bi = bias[0 * HID + ncol];
  const float bf = bias[1 * HID + ncol];
  const float bg = bias[2 * HID + ncol];
  const float bo_ = bias[3 * HID + ncol];
  const int rofs = (lane & 16) ? 8 : 0;
  #pragma unroll
  for (int r = 0; r < 8; ++r) {
    const int b = b0 + r + rofs;
    const float ig = sigmoidf_(acc[0][r] + bi);
    const float fg = sigmoidf_(acc[1][r] + bf);
    const float mg = tanhf    (acc[2][r] + bg);
    const float og = sigmoidf_(acc[3][r] + bo_);
    const size_t ix = (size_t)b * HID + ncol;
    const float cn = mg * ig + fg * Cst[ix];
    Cst[ix]  = cn;
    Hnew[ix] = og * tanhf(cn);
  }
}

// ---------------------------------------------------------------------------
// One-time (per launch) weight conversion: fp32 -> bf16 hi/lo with K-concat:
//   Wc1[n][0:768] = W1, Wc1[n][768:1280] = R1   (n = g*512 + h)
//   Wc2[n][0:512] = W2, Wc2[n][512:1024] = R2
// ---------------------------------------------------------------------------
__global__ void prep_weights(const float* __restrict__ W1, const float* __restrict__ R1,
                             const float* __restrict__ W2, const float* __restrict__ R2,
                             __bf16* __restrict__ Wc1h, __bf16* __restrict__ Wc1l,
                             __bf16* __restrict__ Wc2h, __bf16* __restrict__ Wc2l)
{
  const int K1 = DIN + HID, K2 = 2 * HID;
  const int total1 = 4 * HID * K1;
  const int total2 = 4 * HID * K2;
  for (int idx = blockIdx.x * blockDim.x + threadIdx.x; idx < total1 + total2;
       idx += gridDim.x * blockDim.x) {
    float v; __bf16 *ph, *pl; int o;
    if (idx < total1) {
      const int n = idx / K1, k = idx - n * K1;
      v = (k < DIN) ? W1[(size_t)n * DIN + k] : R1[(size_t)n * HID + (k - DIN)];
      ph = Wc1h; pl = Wc1l; o = idx;
    } else {
      const int j = idx - total1;
      const int n = j / K2, k = j - n * K2;
      v = (k < HID) ? W2[(size_t)n * HID + k] : R2[(size_t)n * HID + (k - HID)];
      ph = Wc2h; pl = Wc2l; o = j;
    }
    const __bf16 hi = (__bf16)v;
    ph[o] = hi;
    pl[o] = (__bf16)(v - (float)hi);
  }
}

__global__ void zero_f32(float* __restrict__ p, int n) {
  for (int i = blockIdx.x * blockDim.x + threadIdx.x; i < n; i += gridDim.x * blockDim.x)
    p[i] = 0.f;
}

// out[b][t][c] = h2[b] . Wo[c] + bo[c]   (tiny 256x9 GEMV per step)
__global__ __launch_bounds__(BATCH) void logits_kernel(
    const float* __restrict__ h2, const float* __restrict__ Wo,
    const float* __restrict__ bo, float* __restrict__ out, int t)
{
  const int b = threadIdx.x;   // 0..255
  const int c = blockIdx.x;    // 0..8
  const float4* hp = (const float4*)(h2 + (size_t)b * HID);
  const float4* wp = (const float4*)(Wo + (size_t)c * HID);
  float acc = 0.f;
  #pragma unroll 4
  for (int k = 0; k < HID / 4; ++k) {
    const float4 hv = hp[k], wv = wp[k];
    acc += hv.x * wv.x + hv.y * wv.y + hv.z * wv.z + hv.w * wv.w;
  }
  out[((size_t)b * SEQL + t) * NCLS + c] = acc + bo[c];
}

// ---------------------------------------------------------------------------
extern "C" void kernel_launch(void* const* d_in, const int* in_sizes, int n_in,
                              void* d_out, int out_size, void* d_ws, size_t ws_size,
                              hipStream_t stream)
{
  const int*   ids = (const int*)  d_in[0];
  const float* tok = (const float*)d_in[1];
  const float* pos = (const float*)d_in[2];
  const float* W1  = (const float*)d_in[3];
  const float* R1  = (const float*)d_in[4];
  const float* b1  = (const float*)d_in[5];
  const float* W2  = (const float*)d_in[6];
  const float* R2  = (const float*)d_in[7];
  const float* b2  = (const float*)d_in[8];
  const float* Wo  = (const float*)d_in[9];
  const float* bo  = (const float*)d_in[10];
  float* out = (float*)d_out;

  // workspace layout (~22 MB): states (parity double-buffered h) + bf16 weights
  float* st = (float*)d_ws;
  const size_t SH = (size_t)BATCH * HID;           // 131072 floats
  float* h1a = st;             float* h1b = st + SH;
  float* c1  = st + 2 * SH;
  float* h2a = st + 3 * SH;    float* h2b = st + 4 * SH;
  float* c2  = st + 5 * SH;
  __bf16* wbase = (__bf16*)(st + 6 * SH);
  const size_t N1 = (size_t)4 * HID * (DIN + HID); // 2048*1280
  const size_t N2 = (size_t)4 * HID * (2 * HID);   // 2048*1024
  __bf16* Wc1h = wbase;
  __bf16* Wc1l = Wc1h + N1;
  __bf16* Wc2h = Wc1l + N1;
  __bf16* Wc2l = Wc2h + N2;

  prep_weights<<<1024, 256, 0, stream>>>(W1, R1, W2, R2, Wc1h, Wc1l, Wc2h, Wc2l);
  zero_f32<<<512, 256, 0, stream>>>(st, (int)(6 * SH));

  const dim3 cgrid(BATCH / 16, HID / 64);          // 16 x 8 = 128 WGs, 512 waves
  for (int t = 0; t < SEQL; ++t) {
    float* h1p = (t & 1) ? h1b : h1a;
    float* h1n = (t & 1) ? h1a : h1b;
    float* h2p = (t & 1) ? h2b : h2a;
    float* h2n = (t & 1) ? h2a : h2b;
    lstm_cell<<<cgrid, 128, 0, stream>>>(0, t, ids, tok, pos,
                                         nullptr, h1p, Wc1h, Wc1l, b1, c1, h1n, DIN + HID);
    lstm_cell<<<cgrid, 128, 0, stream>>>(1, t, nullptr, nullptr, nullptr,
                                         h1n, h2p, Wc2h, Wc2l, b2, c2, h2n, 2 * HID);
    logits_kernel<<<NCLS, BATCH, 0, stream>>>(h2n, Wo, bo, out, t);
  }
  (void)in_sizes; (void)n_in; (void)out_size; (void)ws_size;
}
